// Attention_20830591385929
// MI455X (gfx1250) — matrix operand order repaired and checked by reading, not compiled, not run on hardware
//
#include <hip/hip_runtime.h>

typedef __attribute__((ext_vector_type(16))) _Float16 v16h;
typedef __attribute__((ext_vector_type(8)))  _Float16 v8h;
typedef __attribute__((ext_vector_type(8)))  float    v8f;
typedef __attribute__((ext_vector_type(4)))  unsigned int u32x4;
typedef __attribute__((ext_vector_type(8)))  int          i32x8;
typedef __attribute__((ext_vector_type(4)))  int          i32x4;

#define BATCH 4
#define LQ    2048
#define LK    2048
#define DMOD  1024
#define DKV   128            // DK == DV == 128
#define MTOT  (BATCH * LQ)   // 8192 flattened rows for the projections
#define NSTEP (LK / 32)      // 64 key steps of 32 rows

#if defined(__gfx1250__) && __has_builtin(__builtin_amdgcn_tensor_load_to_lds) && \
    __has_builtin(__builtin_amdgcn_s_wait_tensorcnt)
#define USE_TDM 1
#else
#define USE_TDM 0
#endif

static __device__ __forceinline__ v8f wmma_f16(v16h a, v16h b, v8f c) {
  // D = A(16x32 f16) * B(32x16 f16) + C(16x16 f32)
  return __builtin_amdgcn_wmma_f32_16x16x32_f16(false, a, false, b, (short)0, c,
                                                false, false);
}

#if USE_TDM
// Issue one 2D TENSOR_LOAD_TO_LDS per the CDNA5 D# layout (CDNA5 ISA):
// group0: count=1 | lds_addr[63:32] | global_addr[120:64] | type=2[127:126]
// group1: data_size=2B | tensor_dim0/1 | tile_dim0/1 | tensor_dim0_stride
// Dims/strides in elements (2-byte), addresses in bytes. Groups 2/3 unused (2D).
// 6-arg builtin: (u32x4 g0, i32x8 g1, i32x4 g2, i32x4 g3, i32x8 extra, i32 cpol)
static __device__ __forceinline__ void tdm_load_2d(unsigned lds_off,
                                                   unsigned long long gaddr,
                                                   unsigned tensor_d0,
                                                   unsigned tensor_d1,
                                                   unsigned long long stride0,
                                                   unsigned tile_d0,
                                                   unsigned tile_d1) {
  u32x4 g0;
  g0[0] = 1u;                                               // count = 1
  g0[1] = lds_off;                                          // LDS byte address
  g0[2] = (unsigned)gaddr;                                  // addr[31:0]
  g0[3] = ((unsigned)(gaddr >> 32) & 0x01FFFFFFu) | 0x80000000u; // addr[56:32] | type=2
  i32x8 g1;
  g1[0] = (int)(1u << 16);                                  // data_size = 1 (2 bytes)
  g1[1] = (int)((tensor_d0 & 0xFFFFu) << 16);               // tensor_dim0[15:0]
  g1[2] = (int)((tensor_d0 >> 16) | ((tensor_d1 & 0xFFFFu) << 16));
  g1[3] = (int)((tensor_d1 >> 16) | (tile_d0 << 16));       // tile_dim0
  g1[4] = (int)(tile_d1 & 0xFFFFu);                         // tile_dim1, tile_dim2=0
  g1[5] = (int)(unsigned)(stride0 & 0xFFFFFFFFu);           // tensor_dim0_stride
  g1[6] = (int)(unsigned)(stride0 >> 32);                   // stride hi, stride1=0
  g1[7] = 0;
  i32x4 z4 = {0, 0, 0, 0};
  i32x8 z8 = {0, 0, 0, 0, 0, 0, 0, 0};
  __builtin_amdgcn_tensor_load_to_lds(g0, g1, z4, z4, z8, 0);
}
#endif

// ---------------------------------------------------------------------------
// Projection GEMM: out = X[M,1024] * W[1024,128] + bias, result stored f16.
// Block = 256 threads (8 waves). Wave w owns the 16-col tile n0 = 16*w.
// blockIdx.x picks the 16-row M strip. TR=1 stores out as [128][M] (for V^T).
// ---------------------------------------------------------------------------
template <int TR>
__global__ __launch_bounds__(256)
void proj_kernel(const float* __restrict__ X, const float* __restrict__ W,
                 const float* __restrict__ bias, _Float16* __restrict__ out) {
  const int lane = threadIdx.x & 31;
  const int wave = threadIdx.x >> 5;
  const int hi   = (lane >= 16);
  const int lcol = lane & 15;
  const int m0   = blockIdx.x * 16;
  const int n0   = wave * 16;

  v8f acc = {};
  for (int c = 0; c < DMOD / 32; ++c) {
    // A fragment (16x32 f16): row = m0+lcol ; k = c*32 + (hi?8:0) + {0..7,16..23}
    const float* arow = X + (size_t)(m0 + lcol) * DMOD + c * 32 + (hi ? 8 : 0);
    v16h a;
#pragma unroll
    for (int e = 0; e < 8; ++e) {
      a[e]     = (_Float16)arow[e];
      a[e + 8] = (_Float16)arow[e + 16];
    }
    // B fragment (32x16 f16): col = n0+lcol ; k = c*32 + (hi?8:0) + {0..7,16..23}
    const float* wcol = W + (size_t)(c * 32 + (hi ? 8 : 0)) * DKV + n0 + lcol;
    v16h bf;
#pragma unroll
    for (int e = 0; e < 16; ++e) bf[e] = (_Float16)wcol[(size_t)(e + (e & 8)) * DKV];

    acc = wmma_f16(a, bf, acc);
  }

  const float bn = bias[n0 + lcol];
#pragma unroll
  for (int j = 0; j < 8; ++j) {
    const int m = m0 + j + (hi ? 8 : 0);   // C layout: M = j + 8*hi, N = lcol
    const int n = n0 + lcol;
    const _Float16 val = (_Float16)(acc[j] + bn);
    if (TR) out[(size_t)n * MTOT + m] = val;   // V^T: [128][8192]
    else    out[(size_t)m * DKV  + n] = val;   // q,k: [8192][128]
  }
}

// ---------------------------------------------------------------------------
// Flash attention. One wave = one 16-row q tile; block = 128 thr (4 waves)
// sharing per-step K/V tiles. Tiles are staged into double-buffered LDS by the
// Tensor Data Mover: the TDM load for step s+1 is issued before computing on
// step s, and TENSORcnt is only drained right before the end-of-step barrier,
// overlapping DMA with the WMMA chain.
// ---------------------------------------------------------------------------
__global__ __launch_bounds__(128)
void attn_kernel(const _Float16* __restrict__ qh, const _Float16* __restrict__ kh,
                 const _Float16* __restrict__ vt, float* __restrict__ out) {
  const int lane = threadIdx.x & 31;
  const int wave = threadIdx.x >> 5;
  const int hi   = (lane >= 16);
  const int lcol = lane & 15;
  const int b    = blockIdx.y;
  const int qt   = blockIdx.x * 4 + wave;           // q tile index in [0,128)
  const size_t qrow = (size_t)b * LQ + qt * 16 + lcol;

  __shared__ __align__(16) _Float16 ktile[2][32 * DKV];   // [buf][keyrow][dk] 2x8KB
  __shared__ __align__(16) _Float16 vtile[2][DKV * 32];   // [buf][dv][keyrow] 2x8KB

  const float rs = 0.08838834764831845f;            // 1/sqrt(DK)

  // q^T B fragments, one per 32-wide dk chunk (scale folded in here).
  v16h qb[4];
#pragma unroll
  for (int c = 0; c < 4; ++c) {
    const _Float16* p = qh + qrow * DKV + c * 32 + (hi ? 8 : 0);
    const v8h q0 = *(const v8h*)p;
    const v8h q1 = *(const v8h*)(p + 16);
#pragma unroll
    for (int e = 0; e < 8; ++e) {
      qb[c][e]     = (_Float16)((float)q0[e] * rs);
      qb[c][e + 8] = (_Float16)((float)q1[e] * rs);
    }
  }

  // ---- tile staging (TDM or cooperative fallback) ----
  auto stage = [&](int buf, int k0) {
#if USE_TDM
    if (wave == 0) {
      tdm_load_2d((unsigned)(unsigned long long)&ktile[buf][0],
                  (unsigned long long)(const void*)(kh + ((size_t)b * LK + k0) * DKV),
                  /*tensor_d0=*/DKV, /*tensor_d1=*/MTOT, /*stride0=*/DKV,
                  /*tile_d0=*/DKV, /*tile_d1=*/32);
      tdm_load_2d((unsigned)(unsigned long long)&vtile[buf][0],
                  (unsigned long long)(const void*)(vt + (size_t)b * LK + k0),
                  /*tensor_d0=*/MTOT, /*tensor_d1=*/DKV, /*stride0=*/MTOT,
                  /*tile_d0=*/32, /*tile_d1=*/DKV);
    }
#else
    const v8h* ksrc = (const v8h*)(kh + ((size_t)b * LK + k0) * DKV);
    v8h* kdst = (v8h*)&ktile[buf][0];
    for (int i = threadIdx.x; i < (32 * DKV) / 8; i += 128) kdst[i] = ksrc[i];
    v8h* vdst = (v8h*)&vtile[buf][0];
    for (int i = threadIdx.x; i < (DKV * 32) / 8; i += 128) {
      const int dv = i >> 2, xo = (i & 3) * 8;
      vdst[i] = *(const v8h*)(vt + (size_t)dv * MTOT + (size_t)b * LK + k0 + xo);
    }
#endif
  };

  v8f o[8] = {};                 // O^T accumulators: dv tile t, q col = lcol
  float mrun = -INFINITY, lrun = 0.f;

  // prologue: fill buffer 0
  stage(0, 0);
#if USE_TDM
  if (wave == 0) __builtin_amdgcn_s_wait_tensorcnt(0);
#endif
  __syncthreads();

  for (int s = 0; s < NSTEP; ++s) {
    const int cur = s & 1;

    // kick off the DMA for the next tile into the other buffer (no wait yet)
    if (s + 1 < NSTEP) stage(1 - cur, (s + 1) * 32);

    // ---- load all 8 K A-fragments, then chain 8 WMMAs for S^T ----
    const _Float16* kt = &ktile[cur][0];
    v16h ka[8];                          // [0..3]: keyrows 0-15, [4..7]: 16-31
#pragma unroll
    for (int c = 0; c < 4; ++c) {
      const _Float16* kr0 = kt + (unsigned)lcol * DKV + c * 32 + (hi ? 8 : 0);
      const _Float16* kr1 = kr0 + 16 * DKV;
      const v8h a0 = *(const v8h*)kr0, a1 = *(const v8h*)(kr0 + 16);
      const v8h b0 = *(const v8h*)kr1, b1 = *(const v8h*)(kr1 + 16);
#pragma unroll
      for (int e = 0; e < 8; ++e) {
        ka[c][e]     = a0[e]; ka[c][e + 8]     = a1[e];
        ka[4 + c][e] = b0[e]; ka[4 + c][e + 8] = b1[e];
      }
    }
    v8f st0 = {}, st1 = {};
#pragma unroll
    for (int c = 0; c < 4; ++c) {
      st0 = wmma_f16(ka[c],     qb[c], st0);   // keyrows k0..k0+15
      st1 = wmma_f16(ka[4 + c], qb[c], st1);   // keyrows k0+16..k0+31
    }

    // ---- online softmax (per q column lcol; partner lane is lane^16) ----
    float mstep = -INFINITY;
#pragma unroll
    for (int j = 0; j < 8; ++j) mstep = fmaxf(mstep, fmaxf(st0[j], st1[j]));
    mstep = fmaxf(mstep, __shfl_xor(mstep, 16, 32));
    const float mnew  = fmaxf(mrun, mstep);
    const float alpha = __expf(mrun - mnew);

    v8f p0, p1;
    float psum = 0.f;
#pragma unroll
    for (int j = 0; j < 8; ++j) {
      p0[j] = __expf(st0[j] - mnew);
      p1[j] = __expf(st1[j] - mnew);
      psum += p0[j] + p1[j];
    }
    psum += __shfl_xor(psum, 16, 32);
    lrun = lrun * alpha + psum;
    mrun = mnew;

#pragma unroll
    for (int t = 0; t < 8; ++t)
#pragma unroll
      for (int j = 0; j < 8; ++j) o[t][j] *= alpha;

    // ---- P^T as f16 B fragment: C-layout already matches B-layout, own p0/p1 used as is ----
    v16h pb;
#pragma unroll
    for (int e = 0; e < 8; ++e) {
      const float x0 = __shfl_xor(p0[e], 16, 32);   // partner's p0[e]
      const float x1 = __shfl_xor(p1[e], 16, 32);   // partner's p1[e]
      pb[e]     = (_Float16)p0[e];                  // keyrow (hi?8:0)+e
      pb[e + 8] = (_Float16)p1[e];                  // keyrow 16+(hi?8:0)+e
    }

    // ---- O^T += V^T * P^T : 8 WMMAs over dv tiles (loads paired ahead) ----
    const _Float16* vtb = &vtile[cur][0];
#pragma unroll
    for (int t = 0; t < 8; t += 2) {
      const _Float16* vp0 = vtb + (unsigned)(t * 16 + lcol) * 32 + (hi ? 8 : 0);
      const _Float16* vp1 = vp0 + 16 * 32;
      const v8h l0 = *(const v8h*)vp0, h0 = *(const v8h*)(vp0 + 16);
      const v8h l1 = *(const v8h*)vp1, h1 = *(const v8h*)(vp1 + 16);
      v16h va0, va1;
#pragma unroll
      for (int e = 0; e < 8; ++e) {
        va0[e] = l0[e]; va0[e + 8] = h0[e];
        va1[e] = l1[e]; va1[e + 8] = h1[e];
      }
      o[t]     = wmma_f16(va0, pb, o[t]);
      o[t + 1] = wmma_f16(va1, pb, o[t + 1]);
    }

    // drain the in-flight DMA, then release both buffers for the next step
#if USE_TDM
    if (wave == 0) __builtin_amdgcn_s_wait_tensorcnt(0);
#endif
    __syncthreads();
  }

  // ---- normalize and store: out[b][q][dv] fp32 ----
  const float inv = 1.f / lrun;
  const size_t obase = ((size_t)b * LQ + qt * 16 + lcol) * DKV;
#pragma unroll
  for (int t = 0; t < 8; ++t)
#pragma unroll
    for (int j = 0; j < 8; ++j)
      out[obase + t * 16 + j + (hi ? 8 : 0)] = o[t][j] * inv;
}

// ---------------------------------------------------------------------------
extern "C" void kernel_launch(void* const* d_in, const int* in_sizes, int n_in,
                              void* d_out, int out_size, void* d_ws, size_t ws_size,
                              hipStream_t stream) {
  const float* Q  = (const float*)d_in[0];
  const float* K  = (const float*)d_in[1];
  const float* V  = (const float*)d_in[2];
  const float* WQ = (const float*)d_in[3];
  const float* bQ = (const float*)d_in[4];
  const float* WK = (const float*)d_in[5];
  const float* bK = (const float*)d_in[6];
  const float* WV = (const float*)d_in[7];
  const float* bV = (const float*)d_in[8];

  // workspace: q (2MB) | k (2MB) | v^T (2MB), all f16
  const size_t elems = (size_t)MTOT * DKV;
  _Float16* qh = (_Float16*)d_ws;
  _Float16* kh = qh + elems;
  _Float16* vt = kh + elems;

  dim3 pgrid(MTOT / 16);                       // 512 blocks
  proj_kernel<0><<<pgrid, 256, 0, stream>>>(Q, WQ, bQ, qh);
  proj_kernel<0><<<pgrid, 256, 0, stream>>>(K, WK, bK, kh);
  proj_kernel<1><<<pgrid, 256, 0, stream>>>(V, WV, bV, vt);

  dim3 agrid(LQ / 16 / 4, BATCH);              // (32, 4), 4 waves per block
  attn_kernel<<<agrid, 128, 0, stream>>>(qh, kh, vt, (float*)d_out);
}
